// SelfAttention_36155034698495
// MI455X (gfx1250) — matrix operand order repaired and checked by reading, not compiled, not run on hardware
//
#include <hip/hip_runtime.h>
#include <hip/hip_bf16.h>
#include <stdint.h>

typedef _Float16 half_t;
typedef __attribute__((ext_vector_type(16))) _Float16 v16h;
typedef __attribute__((ext_vector_type(8)))  _Float16 v8h;
typedef __attribute__((ext_vector_type(4)))  _Float16 v4h;
typedef __attribute__((ext_vector_type(8)))  float    v8f;

#define B_SZ 4
#define SEQ 2048
#define DIN 1024
#define DM  1024
#define MTOT (B_SZ * SEQ)   // 8192

// ---------------- fragment loaders ----------------
// A fragment, 16x32 f16, row-major source. lane holds row m = lane&15,
// K = 16*half + j (j<16). Two contiguous 16B loads.
static __device__ inline v16h load_a_frag(const half_t* __restrict__ p, int lh) {
    v8h lo = *(const v8h*)(p + lh * 16);
    v8h hi = *(const v8h*)(p + lh * 16 + 8);
    return __builtin_shufflevector(lo, hi, 0,1,2,3,4,5,6,7,8,9,10,11,12,13,14,15);
}
// B fragment, 32x16 f16, from Bt row-major [N][K]: lane holds col n = lane&15,
// K = 16*half .. 16*half+15 -> one contiguous 32B load per lane.
static __device__ inline v16h load_b_frag(const half_t* __restrict__ prow, int kk) {
    return *(const v16h*)(prow + kk);
}

// ---------------- unified WMMA GEMM ----------------
// C[M,N] = alpha * A[M,K] * Bt[N,K]^T  (+ epilogue per mode)
// mode 0: f16 out row-major, + bias[n] if bias
// mode 1: f16 out transposed C[n*ldc+m], + bias[n] if bias   (for V^T)
// mode 2: f16 out row-major, val = alpha*acc + mask[m*ldc+n]*(-3e4f)  (scores)
// mode 3: f32 out row-major, + bias[n]                        (final output)
// grid: (M/64, N/256, batches); block: 256 threads = 8 waves (2 M-waves x 4 N-waves)
__global__ __launch_bounds__(256)
void wmma_gemm_kernel(const half_t* __restrict__ A, long long sA,
                      const half_t* __restrict__ Bt, long long sB,
                      void* __restrict__ C, long long sC,
                      const float* __restrict__ bias,
                      const int* __restrict__ mask, long long sMask,
                      int Kd, int lda, int ldb, int ldc,
                      float alpha, int mode) {
    const int lane = threadIdx.x & 31;
    const int wid  = threadIdx.x >> 5;
    const int lr   = lane & 15;
    const int lh   = lane >> 4;

    const int m0 = blockIdx.x * 64  + (wid & 1) * 32;
    const int n0 = blockIdx.y * 256 + (wid >> 1) * 64;

    A  += (long long)blockIdx.z * sA;
    Bt += (long long)blockIdx.z * sB;

    const half_t* pa0 = A + (long long)(m0 + lr) * lda;
    const half_t* pa1 = A + (long long)(m0 + 16 + lr) * lda;
    const half_t* pb[4];
#pragma unroll
    for (int s = 0; s < 4; ++s)
        pb[s] = Bt + (long long)(n0 + 16 * s + lr) * ldb + lh * 16;

    v8f acc[2][4] = {};

    for (int kk = 0; kk < Kd; kk += 32) {
        // prefetch ahead on the streaming operands (global_prefetch_b8)
        __builtin_prefetch(pa0 + kk + 256, 0, 1);
        __builtin_prefetch(pb[0] + kk + 256, 0, 1);

        v16h a0 = load_a_frag(pa0 + kk, lh);
        v16h a1 = load_a_frag(pa1 + kk, lh);
#pragma unroll
        for (int s = 0; s < 4; ++s) {
            v16h b = load_b_frag(pb[s], kk);
            acc[0][s] = __builtin_amdgcn_wmma_f32_16x16x32_f16(
                false, a0, false, b, (short)0, acc[0][s], false, false);
            acc[1][s] = __builtin_amdgcn_wmma_f32_16x16x32_f16(
                false, a1, false, b, (short)0, acc[1][s], false, false);
        }
    }

    // ---------------- epilogue ----------------
    if (mode == 2) {
        half_t* Cc = (half_t*)C + (long long)blockIdx.z * sC;
        const int* mk = mask + (long long)blockIdx.z * sMask;
#pragma unroll
        for (int i = 0; i < 2; ++i)
#pragma unroll
            for (int s = 0; s < 4; ++s) {
                const int n = n0 + 16 * s + lr;
#pragma unroll
                for (int j = 0; j < 8; ++j) {
                    const int m = m0 + 16 * i + 8 * lh + j;
                    float v = acc[i][s][j] * alpha +
                              (float)mk[(long long)m * ldc + n] * (-30000.0f);
                    Cc[(long long)m * ldc + n] = (half_t)v;
                }
            }
    } else if (mode == 1) {
        half_t* Cc = (half_t*)C + (long long)blockIdx.z * sC;
#pragma unroll
        for (int i = 0; i < 2; ++i)
#pragma unroll
            for (int s = 0; s < 4; ++s) {
                const int n = n0 + 16 * s + lr;
                const float bv = bias ? bias[n] : 0.0f;
#pragma unroll
                for (int j = 0; j < 8; ++j) {
                    const int m = m0 + 16 * i + 8 * lh + j;
                    Cc[(long long)n * ldc + m] = (half_t)(acc[i][s][j] * alpha + bv);
                }
            }
    } else if (mode == 3) {
        float* Cf = (float*)C + (long long)blockIdx.z * sC;
#pragma unroll
        for (int i = 0; i < 2; ++i)
#pragma unroll
            for (int s = 0; s < 4; ++s) {
                const int n = n0 + 16 * s + lr;
                const float bv = bias ? bias[n] : 0.0f;
#pragma unroll
                for (int j = 0; j < 8; ++j) {
                    const int m = m0 + 16 * i + 8 * lh + j;
                    Cf[(long long)m * ldc + n] = acc[i][s][j] * alpha + bv;
                }
            }
    } else { // mode 0
        half_t* Cc = (half_t*)C + (long long)blockIdx.z * sC;
#pragma unroll
        for (int i = 0; i < 2; ++i)
#pragma unroll
            for (int s = 0; s < 4; ++s) {
                const int n = n0 + 16 * s + lr;
                const float bv = bias ? bias[n] : 0.0f;
#pragma unroll
                for (int j = 0; j < 8; ++j) {
                    const int m = m0 + 16 * i + 8 * lh + j;
                    Cc[(long long)m * ldc + n] = (half_t)(acc[i][s][j] * alpha + bv);
                }
            }
    }
}

// ---------------- fp32 -> f16 convert (x4 vectorized) ----------------
__global__ void cvt_f32_f16_x4(const float* __restrict__ in,
                               half_t* __restrict__ out, long long n4) {
    long long i = (long long)blockIdx.x * blockDim.x + threadIdx.x;
    if (i < n4) {
        float4 v = ((const float4*)in)[i];
        v4h h = { (half_t)v.x, (half_t)v.y, (half_t)v.z, (half_t)v.w };
        ((v4h*)out)[i] = h;
    }
}

// ---------------- fp32 -> f16 transpose (LDS tiled) ----------------
// out[c*rows + r] = (half)in[r*cols + c] ; grid (cols/32, rows/32), block (32,8)
__global__ void transpose_cvt_f32_f16(const float* __restrict__ in,
                                      half_t* __restrict__ out,
                                      int rows, int cols) {
    __shared__ float tile[32][33];
    const int bx = blockIdx.x * 32;  // col base
    const int by = blockIdx.y * 32;  // row base
    const int tx = threadIdx.x, ty = threadIdx.y;
#pragma unroll
    for (int j = 0; j < 32; j += 8)
        tile[ty + j][tx] = in[(long long)(by + ty + j) * cols + bx + tx];
    __syncthreads();
#pragma unroll
    for (int j = 0; j < 32; j += 8)
        out[(long long)(bx + ty + j) * rows + by + tx] = (half_t)tile[tx][ty + j];
}

// ---------------- row softmax, in place on f16 [rows][2048] ----------------
// gfx1250 async path: stage the 4KB row Global->LDS with
// global_load_async_to_lds_b128 (ASYNCcnt), compute from LDS, write results
// back to LDS and drain with global_store_async_from_lds_b128.
__global__ __launch_bounds__(256)
void softmax_rows_f16(half_t* __restrict__ S, int ncols) {
    __shared__ half_t srow[SEQ];          // 4 KB row staging
    __shared__ float red_m[8];
    __shared__ float red_s[8];

    half_t* p = S + (long long)blockIdx.x * ncols;
    const int tid = threadIdx.x;

    const half_t* gsrc   = p + tid * 8;                       // 16 B per lane
    half_t*       gdst   = p + tid * 8;
    unsigned      ldsoff = (unsigned)(uintptr_t)&srow[tid * 8];

    // ---- async copy Global -> LDS (tracked by ASYNCcnt) ----
    asm volatile("global_load_async_to_lds_b128 %0, %1, off"
                 :: "v"(ldsoff), "v"(gsrc) : "memory");
    asm volatile("s_wait_asynccnt 0" ::: "memory");
    __syncthreads();

    // each thread owns 8 contiguous elements: one ds_load_b128
    float x[8];
    {
        v8h raw = *(const v8h*)&srow[tid * 8];
#pragma unroll
        for (int j = 0; j < 8; ++j) x[j] = (float)raw[j];
    }

    float m = -3.4e38f;
#pragma unroll
    for (int j = 0; j < 8; ++j) m = fmaxf(m, x[j]);
#pragma unroll
    for (int off = 16; off >= 1; off >>= 1)
        m = fmaxf(m, __shfl_xor(m, off, 32));
    if ((tid & 31) == 0) red_m[tid >> 5] = m;
    __syncthreads();   // also: all LDS row reads complete before overwrite below
#pragma unroll
    for (int w = 0; w < 8; ++w) m = fmaxf(m, red_m[w]);

    float e[8], sum = 0.0f;
#pragma unroll
    for (int j = 0; j < 8; ++j) { e[j] = __expf(x[j] - m); sum += e[j]; }
#pragma unroll
    for (int off = 16; off >= 1; off >>= 1)
        sum += __shfl_xor(sum, off, 32);
    if ((tid & 31) == 0) red_s[tid >> 5] = sum;
    __syncthreads();
    sum = 0.0f;
#pragma unroll
    for (int w = 0; w < 8; ++w) sum += red_s[w];

    const float inv = 1.0f / sum;
    v8h r;
#pragma unroll
    for (int j = 0; j < 8; ++j) r[j] = (half_t)(e[j] * inv);
    *(v8h*)&srow[tid * 8] = r;            // ds_store_b128

    // order DS write before the async engine reads LDS, then LDS -> Global
    asm volatile("s_wait_dscnt 0" ::: "memory");
    asm volatile("global_store_async_from_lds_b128 %0, %1, off"
                 :: "v"(gdst), "v"(ldsoff) : "memory");
    asm volatile("s_wait_asynccnt 0" ::: "memory");
}

// ---------------- workspace layout (bytes) ----------------
#define OFF_XH   0LL                       // 8192*1024 f16 = 16 MiB (reused as O)
#define OFF_WQT  16777216LL                // 1024*1024 f16 each
#define OFF_WKT  18874368LL
#define OFF_WVT  20971520LL
#define OFF_WOT  23068672LL
#define OFF_Q    25165824LL                // 8192*1024 f16
#define OFF_K    41943040LL                // 8192*1024 f16
#define OFF_VT   58720256LL                // 1024*8192 f16 (V transposed, cols b-major)
#define OFF_S    75497472LL                // 4*2048*2048 f16 (scores -> P in place)

extern "C" void kernel_launch(void* const* d_in, const int* in_sizes, int n_in,
                              void* d_out, int out_size, void* d_ws, size_t ws_size,
                              hipStream_t stream) {
    const float* x    = (const float*)d_in[0];
    const int*   mask = (const int*)  d_in[1];
    const float* wq   = (const float*)d_in[2];
    const float* bq   = (const float*)d_in[3];
    const float* wk   = (const float*)d_in[4];
    const float* bk   = (const float*)d_in[5];
    const float* wv   = (const float*)d_in[6];
    const float* bv   = (const float*)d_in[7];
    const float* wo   = (const float*)d_in[8];
    const float* bo   = (const float*)d_in[9];
    float* out = (float*)d_out;

    char* ws = (char*)d_ws;
    half_t* xh  = (half_t*)(ws + OFF_XH);
    half_t* wqt = (half_t*)(ws + OFF_WQT);
    half_t* wkt = (half_t*)(ws + OFF_WKT);
    half_t* wvt = (half_t*)(ws + OFF_WVT);
    half_t* wot = (half_t*)(ws + OFF_WOT);
    half_t* Qh  = (half_t*)(ws + OFF_Q);
    half_t* Kh  = (half_t*)(ws + OFF_K);
    half_t* Vt  = (half_t*)(ws + OFF_VT);
    half_t* Sp  = (half_t*)(ws + OFF_S);
    half_t* Oh  = xh;  // reuse: xh dead after V projection

    // 1) x -> f16
    cvt_f32_f16_x4<<<dim3((MTOT * DIN / 4 + 255) / 256), dim3(256), 0, stream>>>(
        x, xh, (long long)MTOT * DIN / 4);

    // 2) transpose-convert weights: wt[n][k] = w[k][n]
    transpose_cvt_f32_f16<<<dim3(32, 32), dim3(32, 8), 0, stream>>>(wq, wqt, DIN, DM);
    transpose_cvt_f32_f16<<<dim3(32, 32), dim3(32, 8), 0, stream>>>(wk, wkt, DIN, DM);
    transpose_cvt_f32_f16<<<dim3(32, 32), dim3(32, 8), 0, stream>>>(wv, wvt, DIN, DM);
    transpose_cvt_f32_f16<<<dim3(32, 32), dim3(32, 8), 0, stream>>>(wo, wot, DM, DM);

    // 3) Q = xh * wqt^T ; K = xh * wkt^T   (f16 out, row-major)
    wmma_gemm_kernel<<<dim3(MTOT / 64, DM / 256, 1), dim3(256), 0, stream>>>(
        xh, 0, wqt, 0, Qh, 0, bq, nullptr, 0, DIN, DIN, DIN, DM, 1.0f, 0);
    wmma_gemm_kernel<<<dim3(MTOT / 64, DM / 256, 1), dim3(256), 0, stream>>>(
        xh, 0, wkt, 0, Kh, 0, bk, nullptr, 0, DIN, DIN, DIN, DM, 1.0f, 0);
    // 4) V^T: Vt[d][b*S+s]  (mode 1, ldc = 8192)
    wmma_gemm_kernel<<<dim3(MTOT / 64, DM / 256, 1), dim3(256), 0, stream>>>(
        xh, 0, wvt, 0, Vt, 0, bv, nullptr, 0, DIN, DIN, DIN, MTOT, 1.0f, 1);

    // 5) scores = Q*K^T/32 + mask*(-3e4)  (f16 out, batched over z)
    wmma_gemm_kernel<<<dim3(SEQ / 64, SEQ / 256, B_SZ), dim3(256), 0, stream>>>(
        Qh, (long long)SEQ * DM, Kh, (long long)SEQ * DM,
        Sp, (long long)SEQ * SEQ, nullptr, mask, (long long)SEQ * SEQ,
        DM, DM, DM, SEQ, 0.03125f, 2);

    // 6) softmax in place -> P (f16), via async Global<->LDS staging
    softmax_rows_f16<<<dim3(B_SZ * SEQ), dim3(256), 0, stream>>>(Sp, SEQ);

    // 7) O = P * V  (Bt = Vt, per-batch column offset 2048, ldb = 8192)
    wmma_gemm_kernel<<<dim3(SEQ / 64, DM / 256, B_SZ), dim3(256), 0, stream>>>(
        Sp, (long long)SEQ * SEQ, Vt, (long long)SEQ,
        Oh, (long long)SEQ * DM, nullptr, nullptr, 0,
        SEQ, SEQ, MTOT, DM, 1.0f, 0);

    // 8) out = O * wot^T + bo  (fp32 out)
    wmma_gemm_kernel<<<dim3(MTOT / 64, DM / 256, 1), dim3(256), 0, stream>>>(
        Oh, 0, wot, 0, out, 0, bo, nullptr, 0, DM, DM, DM, DM, 1.0f, 3);
}
